// TemplatePairStackIteration_43130061586673
// MI455X (gfx1250) — matrix operand order repaired and checked by reading, not compiled, not run on hardware
//
#include <hip/hip_runtime.h>
#include <hip/hip_bf16.h>

// Model dims (fixed by reference)
#define BDIM   2
#define NDIM   192
#define CDIM   128
#define HDIM   4
#define ACDIM  32
#define MDIM   (BDIM*NDIM*NDIM)   // 73728 pair rows

typedef __bf16 bf16_t;
typedef __attribute__((ext_vector_type(16))) __bf16 v16bf;
typedef __attribute__((ext_vector_type(8)))  float  v8f;

union Frag { unsigned u[8]; uint4 q[2]; v16bf v; };

__device__ __forceinline__ bf16_t f2bf(float f) {
  unsigned u; __builtin_memcpy(&u, &f, 4);
  unsigned r = u + 0x7FFFu + ((u >> 16) & 1u);   // round-to-nearest-even
  unsigned short h = (unsigned short)(r >> 16);
  bf16_t o; __builtin_memcpy(&o, &h, 2);
  return o;
}
__device__ __forceinline__ float bf2f(bf16_t x) {
  unsigned short h; __builtin_memcpy(&h, &x, 2);
  unsigned u = ((unsigned)h) << 16;
  float f; __builtin_memcpy(&f, &u, 4);
  return f;
}

// ---------------------------------------------------------------------------
// Weight convert + transpose: Wt[n*K + k] = bf16(W[k*Nc + n])
// ---------------------------------------------------------------------------
__global__ void convt_kernel(const float* __restrict__ w, bf16_t* __restrict__ wt,
                             int K, int Nc) {
  int i = blockIdx.x * 256 + threadIdx.x;
  if (i >= K * Nc) return;
  int k = i / Nc, n = i % Nc;
  wt[(size_t)n * K + k] = f2bf(w[i]);
}

// ---------------------------------------------------------------------------
// LayerNorm over C=128, fp32 in -> bf16 out
// ---------------------------------------------------------------------------
__global__ __launch_bounds__(128) void ln_kernel(const float* __restrict__ x,
    const float* __restrict__ g, const float* __restrict__ b,
    bf16_t* __restrict__ out) {
  __shared__ float red[4];
  size_t row = blockIdx.x;
  int c = threadIdx.x;
  float v = x[row * 128 + c];
  float s = v;
  #pragma unroll
  for (int o = 16; o; o >>= 1) s += __shfl_xor(s, o, 32);
  if ((c & 31) == 0) red[c >> 5] = s;
  __syncthreads();
  float mean = (red[0] + red[1] + red[2] + red[3]) * (1.0f / 128.0f);
  float d = v - mean;
  float s2 = d * d;
  #pragma unroll
  for (int o = 16; o; o >>= 1) s2 += __shfl_xor(s2, o, 32);
  __syncthreads();
  if ((c & 31) == 0) red[c >> 5] = s2;
  __syncthreads();
  float var = (red[0] + red[1] + red[2] + red[3]) * (1.0f / 128.0f);
  float r = rsqrtf(var + 1e-5f);
  out[row * 128 + c] = f2bf(d * r * g[c] + b[c]);
}

// ---------------------------------------------------------------------------
// WMMA GEMM: out[M,Nc] = act(A_bf16[M,K] @ W (transposed bf16 [Nc,K]) + bias)
// Block tile 128x128, 8 waves; wave wv owns m-tile wv and all 8 n-tiles.
// A fragment amortized over 8 WMMAs; fragments loaded as 2x b128 from LDS.
// OUT: 0 = f32 store, 1 = bf16 store, 2 = resid[idx] += v * gate[idx] * rowmask[m]
// ---------------------------------------------------------------------------
template<int ACT, int OUTM>
__global__ __launch_bounds__(256) void gemm_kernel(
    const bf16_t* __restrict__ A, const bf16_t* __restrict__ Wt,
    const float* __restrict__ bias,
    float* __restrict__ outF, bf16_t* __restrict__ outH,
    float* __restrict__ resid, const float* __restrict__ gate,
    const float* __restrict__ rowmask,
    int M, int Nc, int K) {
  __shared__ __align__(16) bf16_t Alds[128][32];
  __shared__ __align__(16) bf16_t Blds[128][32];
  const int tid  = threadIdx.x;
  const int lane = tid & 31;
  const int wv   = tid >> 5;       // m-tile index 0..7
  const int bm   = blockIdx.x * 128;
  const int bn   = blockIdx.y * 128;
  const int hh   = lane >> 4;
  const int mr   = lane & 15;
  v8f acc[8];
  #pragma unroll
  for (int n = 0; n < 8; ++n) acc[n] = (v8f){};
  for (int kb = 0; kb < K; kb += 32) {
    __syncthreads();
    #pragma unroll
    for (int s = 0; s < 2; ++s) {
      int idx = tid + s * 256;
      int row = idx >> 2, seg = (idx & 3) * 8;
      *(uint4*)&Alds[row][seg] = *(const uint4*)(A  + (size_t)(bm + row) * K + kb + seg);
      *(uint4*)&Blds[row][seg] = *(const uint4*)(Wt + (size_t)(bn + row) * K + kb + seg);
    }
    if (kb + 32 < K)
      __builtin_prefetch(A + (size_t)(bm + (tid >> 2)) * K + kb + 32, 0, 0);
    __syncthreads();
    Frag a;
    a.q[0] = *(const uint4*)&Alds[wv * 16 + mr][16 * hh];
    a.q[1] = *(const uint4*)&Alds[wv * 16 + mr][16 * hh + 8];
    #pragma unroll
    for (int n = 0; n < 8; ++n) {
      Frag bfr;
      bfr.q[0] = *(const uint4*)&Blds[n * 16 + mr][16 * hh];
      bfr.q[1] = *(const uint4*)&Blds[n * 16 + mr][16 * hh + 8];
      acc[n] = __builtin_amdgcn_wmma_f32_16x16x32_bf16(false, a.v, false, bfr.v,
                                                       (short)0, acc[n], false, false);
    }
  }
  const int mbase = bm + wv * 16 + 8 * hh;
  #pragma unroll
  for (int n = 0; n < 8; ++n) {
    int col = bn + n * 16 + mr;
    float bv = bias ? bias[col] : 0.0f;
    #pragma unroll
    for (int g = 0; g < 8; ++g) {
      int m = mbase + g;
      float v = acc[n][g] + bv;
      if (ACT == 1) v = 1.0f / (1.0f + __expf(-v));
      else if (ACT == 2) v = fmaxf(v, 0.0f);
      size_t idx = (size_t)m * Nc + col;
      if (OUTM == 0)      outF[idx] = v;
      else if (OUTM == 1) outH[idx] = f2bf(v);
      else {
        float gm = gate ? gate[idx] : 1.0f;
        float mk = rowmask ? rowmask[m] : 1.0f;
        resid[idx] += v * gm * mk;
      }
    }
  }
}

// ---------------------------------------------------------------------------
// Bias projection b = xn @ bw  (C=128 -> H=4)
// ---------------------------------------------------------------------------
__global__ void bproj_kernel(const bf16_t* __restrict__ xn,
                             const float* __restrict__ bw,
                             float* __restrict__ bp, int M) {
  int i = blockIdx.x * 256 + threadIdx.x;
  if (i >= M * HDIM) return;
  int row = i >> 2, h = i & 3;
  float s = 0.0f;
  #pragma unroll 4
  for (int c = 0; c < CDIM; ++c)
    s += bf2f(xn[(size_t)row * CDIM + c]) * bw[c * HDIM + h];
  bp[i] = s;
}

// ---------------------------------------------------------------------------
// Fused triangle attention: one workgroup per (b, fixed-idx, head)
// START: rows = j, keys = k, fixed = i ; END: rows = i, keys = k, fixed = j
// ---------------------------------------------------------------------------
template<bool START>
__global__ __launch_bounds__(128) void attn_kernel(
    const bf16_t* __restrict__ qb, const bf16_t* __restrict__ kb,
    const bf16_t* __restrict__ vb, const bf16_t* __restrict__ gb,
    const float* __restrict__ bp, const float* __restrict__ mask,
    bf16_t* __restrict__ ob) {
  __shared__ __align__(16) bf16_t Ql[NDIM][ACDIM];
  __shared__ __align__(16) bf16_t Kl[NDIM][ACDIM];
  __shared__ __align__(16) bf16_t Vt[ACDIM][NDIM];
  __shared__ __align__(16) bf16_t Pl[4][16][NDIM];

  const int bid = blockIdx.x;
  const int h   = bid & 3;
  const int fix = (bid >> 2) % NDIM;
  const int b   = bid / (4 * NDIM);
  const int tid = threadIdx.x;
  const int wv  = tid >> 5;
  const int lane = tid & 31;
  const int nn = lane & 15;
  const int hh = lane >> 4;

  auto rowg = [&](int r) -> size_t {
    return START ? ((size_t)(b * NDIM + fix) * NDIM + r)
                 : ((size_t)(b * NDIM + r) * NDIM + fix);
  };

  for (int e = tid; e < NDIM * ACDIM; e += 128) {
    int r = e >> 5, ac = e & 31;
    size_t gi = rowg(r) * CDIM + ac * HDIM + h;
    Ql[r][ac] = qb[gi];
    Kl[r][ac] = kb[gi];
    Vt[ac][r] = vb[gi];
  }
  __syncthreads();

  const float inv = 0.17677669529663688f;  // 1/sqrt(32)

  for (int jt = wv; jt < NDIM / 16; jt += 4) {
    // ---- logits: Q(16x32) @ K^T -> 12 tiles of 16x16 kept in registers ----
    Frag qa;
    qa.q[0] = *(const uint4*)&Ql[jt * 16 + nn][16 * hh];
    qa.q[1] = *(const uint4*)&Ql[jt * 16 + nn][16 * hh + 8];
    v8f lacc[12];
    #pragma unroll
    for (int t = 0; t < 12; ++t) {
      Frag kf;
      kf.q[0] = *(const uint4*)&Kl[t * 16 + nn][16 * hh];
      kf.q[1] = *(const uint4*)&Kl[t * 16 + nn][16 * hh + 8];
      v8f z = {};
      lacc[t] = __builtin_amdgcn_wmma_f32_16x16x32_bf16(false, qa.v, false, kf.v,
                                                        (short)0, z, false, false);
    }
    // ---- bias + mask ----
    #pragma unroll
    for (int t = 0; t < 12; ++t) {
      int key = t * 16 + nn;
      float mk = START ? mask[(size_t)(b * NDIM + fix) * NDIM + key]
                       : mask[(size_t)(b * NDIM + key) * NDIM + fix];
      #pragma unroll
      for (int g = 0; g < 8; ++g) {
        int rrow = jt * 16 + g + 8 * hh;
        float bias = START ? bp[((size_t)(b * NDIM + key) * NDIM + rrow) * HDIM + h]
                           : bp[((size_t)(b * NDIM + rrow) * NDIM + key) * HDIM + h];
        float v = lacc[t][g] * inv + bias;
        lacc[t][g] = (v + 100.0f) * mk - 100.0f;
      }
    }
    // ---- softmax over keys (row = same lane-half; reduce 16 lanes x 12 tiles) ----
    #pragma unroll
    for (int g = 0; g < 8; ++g) {
      float mx = -3.0e38f;
      #pragma unroll
      for (int t = 0; t < 12; ++t) mx = fmaxf(mx, lacc[t][g]);
      #pragma unroll
      for (int o = 1; o < 16; o <<= 1) mx = fmaxf(mx, __shfl_xor(mx, o, 32));
      float se = 0.0f;
      #pragma unroll
      for (int t = 0; t < 12; ++t) {
        float e = __expf(lacc[t][g] - mx);
        lacc[t][g] = e;
        se += e;
      }
      #pragma unroll
      for (int o = 1; o < 16; o <<= 1) se += __shfl_xor(se, o, 32);
      float rs = 1.0f / se;
      #pragma unroll
      for (int t = 0; t < 12; ++t)
        Pl[wv][g + 8 * hh][t * 16 + nn] = f2bf(lacc[t][g] * rs);
    }
    // ---- O = P(16x192) @ V(192x32), gate, store ----
    #pragma unroll
    for (int acT = 0; acT < 2; ++acT) {
      v8f oacc = {};
      #pragma unroll
      for (int kb2 = 0; kb2 < 6; ++kb2) {
        Frag pa, vf;
        pa.q[0] = *(const uint4*)&Pl[wv][nn][kb2 * 32 + 16 * hh];
        pa.q[1] = *(const uint4*)&Pl[wv][nn][kb2 * 32 + 16 * hh + 8];
        vf.q[0] = *(const uint4*)&Vt[acT * 16 + nn][kb2 * 32 + 16 * hh];
        vf.q[1] = *(const uint4*)&Vt[acT * 16 + nn][kb2 * 32 + 16 * hh + 8];
        oacc = __builtin_amdgcn_wmma_f32_16x16x32_bf16(false, pa.v, false, vf.v,
                                                       (short)0, oacc, false, false);
      }
      #pragma unroll
      for (int g = 0; g < 8; ++g) {
        int rrow = jt * 16 + g + 8 * hh;
        int ac = acT * 16 + nn;
        size_t oi = rowg(rrow) * CDIM + ac * HDIM + h;
        ob[oi] = f2bf(oacc[g] * bf2f(gb[oi]));
      }
    }
  }
}

// ---------------------------------------------------------------------------
// Gated pointwise + transpose to channel-major [b][c][n1][n2] (SWAP: [b][c][n2][n1])
// ---------------------------------------------------------------------------
template<bool SWAP>
__global__ void gatewrite_kernel(const float* __restrict__ a,
                                 const float* __restrict__ s,
                                 const float* __restrict__ mask,
                                 bf16_t* __restrict__ dst) {
  size_t i = (size_t)blockIdx.x * 256 + threadIdx.x;
  if (i >= (size_t)MDIM * CDIM) return;
  int c = i & 127;
  size_t row = i >> 7;
  int n2 = row % NDIM;
  size_t t = row / NDIM;
  int n1 = t % NDIM;
  int b = (int)(t / NDIM);
  float v = a[i] * s[i] * mask[row];
  size_t d = SWAP ? ((((size_t)b * CDIM + c) * NDIM + n2) * NDIM + n1)
                  : ((((size_t)b * CDIM + c) * NDIM + n1) * NDIM + n2);
  dst[d] = f2bf(v);
}

// ---------------------------------------------------------------------------
// Triangle-mult einsum: per (b,c): O[i,j] = sum_k It[i,k] * Jt[j,k]  (NT GEMM)
// Wave pre-loads all 6 A fragments for its i-tile, sweeps 12 j-tiles.
// out laid out [b,i,j,c]
// ---------------------------------------------------------------------------
__global__ __launch_bounds__(128) void trimul_kernel(
    const bf16_t* __restrict__ It, const bf16_t* __restrict__ Jt,
    float* __restrict__ out) {
  int bc = blockIdx.x;           // 0..B*C-1
  int c = bc & 127, b = bc >> 7;
  const bf16_t* Ib = It + (size_t)bc * NDIM * NDIM;
  const bf16_t* Jb = Jt + (size_t)bc * NDIM * NDIM;
  const int wv = threadIdx.x >> 5;
  const int lane = threadIdx.x & 31;
  const int nn = lane & 15, hh = lane >> 4;
  for (int it = wv; it < 12; it += 4) {
    Frag af[6];
    #pragma unroll
    for (int kb = 0; kb < 6; ++kb) {
      const bf16_t* base = Ib + (size_t)(it * 16 + nn) * NDIM + kb * 32;
      af[kb].q[0] = *(const uint4*)(base + 16 * hh);
      af[kb].q[1] = *(const uint4*)(base + 16 * hh + 8);
    }
    for (int jt = 0; jt < 12; ++jt) {
      v8f acc = {};
      #pragma unroll
      for (int kb = 0; kb < 6; ++kb) {
        Frag bfr;
        const bf16_t* base = Jb + (size_t)(jt * 16 + nn) * NDIM + kb * 32;
        bfr.q[0] = *(const uint4*)(base + 16 * hh);
        bfr.q[1] = *(const uint4*)(base + 16 * hh + 8);
        acc = __builtin_amdgcn_wmma_f32_16x16x32_bf16(false, af[kb].v, false, bfr.v,
                                                      (short)0, acc, false, false);
      }
      #pragma unroll
      for (int g = 0; g < 8; ++g) {
        int irow = it * 16 + g + 8 * hh;
        int jcol = jt * 16 + nn;
        out[(((size_t)b * NDIM + irow) * NDIM + jcol) * CDIM + c] = acc[g];
      }
    }
  }
}

__global__ void maskmul_kernel(float* __restrict__ x, const float* __restrict__ mask) {
  size_t i = (size_t)blockIdx.x * 256 + threadIdx.x;
  if (i >= (size_t)MDIM * CDIM) return;
  x[i] *= mask[i >> 7];
}

// ---------------------------------------------------------------------------
// Host side
// ---------------------------------------------------------------------------
enum { ACT_NONE = 0, ACT_SIG = 1, ACT_RELU = 2 };
enum { OUT_F32 = 0, OUT_BF16 = 1, OUT_RESID = 2 };

static void launch_gemm(int act, int outm, const bf16_t* A, const bf16_t* Wt,
                        const float* bias, float* outF, bf16_t* outH, float* resid,
                        const float* gate, const float* rowmask,
                        int M, int Nc, int K, hipStream_t s) {
  dim3 g(M / 128, Nc / 128), blk(256);
#define GCASE(a, o)                                                                     \
  if (act == a && outm == o) {                                                          \
    gemm_kernel<a, o><<<g, blk, 0, s>>>(A, Wt, bias, outF, outH, resid, gate, rowmask,  \
                                        M, Nc, K);                                      \
    return;                                                                             \
  }
  GCASE(0, 0) GCASE(1, 0) GCASE(0, 1) GCASE(1, 1) GCASE(2, 1) GCASE(0, 2)
#undef GCASE
}

extern "C" void kernel_launch(void* const* d_in, const int* in_sizes, int n_in,
                              void* d_out, int out_size, void* d_ws, size_t ws_size,
                              hipStream_t stream) {
  (void)in_sizes; (void)n_in; (void)out_size; (void)ws_size;
  const int M = MDIM;
  const size_t MC = (size_t)M * CDIM;
  const float* mask = (const float*)d_in[1];
  float* x = (float*)d_out;

  // ---- workspace layout ----
  char* ws = (char*)d_ws;
  size_t off = 0;
  auto walloc = [&](size_t bytes) -> char* {
    char* p = ws + off;
    off = (off + bytes + 255) & ~(size_t)255;
    return p;
  };
  bf16_t* xn   = (bf16_t*)walloc(MC * 2);          // LN output
  bf16_t* R1   = (bf16_t*)walloc(4 * MC * 2);      // q/k/v/gate  OR  h1 [M,512]
  bf16_t* buf2 = (bf16_t*)walloc(MC * 2);          // attn-out / It / ln2
  bf16_t* buf3 = (bf16_t*)walloc(MC * 2);          // Jt
  float*  bp   = (float*)walloc((size_t)M * HDIM * 4);
  float*  tmpA = (float*)walloc(MC * 4);
  float*  tmpB = (float*)walloc(MC * 4);
  bf16_t* wts  = (bf16_t*)walloc((size_t)491520 * 2);

  bf16_t* qbuf = R1;
  bf16_t* kbuf = R1 + MC;
  bf16_t* vbuf = R1 + 2 * MC;
  bf16_t* gbuf = R1 + 3 * MC;
  bf16_t* h1   = R1;  // [M, 512] reuse after attention+trimul

  // ---- convert weights (transposed bf16) ----
  bf16_t* wp = wts;
  auto conv = [&](int idx, int K, int Nc) -> bf16_t* {
    bf16_t* dst = wp;
    wp += (size_t)K * Nc;
    int tot = K * Nc;
    convt_kernel<<<(tot + 255) / 256, 256, 0, stream>>>((const float*)d_in[idx], dst, K, Nc);
    return dst;
  };
  // att_start @2: ng nb q k v bw gw gb ow ob ; att_end @12
  bf16_t *aqw[2], *akw[2], *avw[2], *agw[2], *aow[2];
  for (int a = 0; a < 2; ++a) {
    int i0 = 2 + a * 10;
    aqw[a] = conv(i0 + 2, CDIM, CDIM);
    akw[a] = conv(i0 + 3, CDIM, CDIM);
    avw[a] = conv(i0 + 4, CDIM, CDIM);
    agw[a] = conv(i0 + 6, CDIM, CDIM);
    aow[a] = conv(i0 + 8, CDIM, CDIM);
  }
  // tm_out @22, tm_in @38: n1g n1b n2g n2b iw ib jw jb isw isb jsw jsb pw pb sw sb
  bf16_t *tiw[2], *tjw[2], *tisw[2], *tjsw[2], *tpw[2], *tsw[2];
  for (int a = 0; a < 2; ++a) {
    int i0 = 22 + a * 16;
    tiw[a]  = conv(i0 + 4, CDIM, CDIM);
    tjw[a]  = conv(i0 + 6, CDIM, CDIM);
    tisw[a] = conv(i0 + 8, CDIM, CDIM);
    tjsw[a] = conv(i0 + 10, CDIM, CDIM);
    tpw[a]  = conv(i0 + 12, CDIM, CDIM);
    tsw[a]  = conv(i0 + 14, CDIM, CDIM);
  }
  bf16_t* w1t = conv(56, CDIM, CDIM * 4);   // [128,512]
  bf16_t* w2t = conv(58, CDIM * 4, CDIM);   // [512,128]

  // ---- x = x2d ----
  hipMemcpyAsync(x, d_in[0], MC * 4, hipMemcpyDeviceToDevice, stream);

  // ================= attention blocks =================
  for (int a = 0; a < 2; ++a) {
    int i0 = 2 + a * 10;
    const float* ng = (const float*)d_in[i0 + 0];
    const float* nb = (const float*)d_in[i0 + 1];
    const float* bw = (const float*)d_in[i0 + 5];
    const float* gbias = (const float*)d_in[i0 + 7];
    const float* obias = (const float*)d_in[i0 + 9];
    ln_kernel<<<M, 128, 0, stream>>>(x, ng, nb, xn);
    launch_gemm(ACT_NONE, OUT_BF16, xn, aqw[a], nullptr, nullptr, qbuf, nullptr, nullptr, nullptr, M, CDIM, CDIM, stream);
    launch_gemm(ACT_NONE, OUT_BF16, xn, akw[a], nullptr, nullptr, kbuf, nullptr, nullptr, nullptr, M, CDIM, CDIM, stream);
    launch_gemm(ACT_NONE, OUT_BF16, xn, avw[a], nullptr, nullptr, vbuf, nullptr, nullptr, nullptr, M, CDIM, CDIM, stream);
    launch_gemm(ACT_SIG,  OUT_BF16, xn, agw[a], gbias,   nullptr, gbuf, nullptr, nullptr, nullptr, M, CDIM, CDIM, stream);
    bproj_kernel<<<(M * HDIM + 255) / 256, 256, 0, stream>>>(xn, bw, bp, M);
    dim3 ag(BDIM * NDIM * HDIM);
    if (a == 0)
      attn_kernel<true><<<ag, 128, 0, stream>>>(qbuf, kbuf, vbuf, gbuf, bp, mask, buf2);
    else
      attn_kernel<false><<<ag, 128, 0, stream>>>(qbuf, kbuf, vbuf, gbuf, bp, mask, buf2);
    launch_gemm(ACT_NONE, OUT_RESID, buf2, aow[a], obias, nullptr, nullptr, x, nullptr, mask, M, CDIM, CDIM, stream);
  }

  // ================= triangle multiplication blocks =================
  for (int a = 0; a < 2; ++a) {
    int i0 = 22 + a * 16;
    const float* n1g = (const float*)d_in[i0 + 0];
    const float* n1b = (const float*)d_in[i0 + 1];
    const float* n2g = (const float*)d_in[i0 + 2];
    const float* n2b = (const float*)d_in[i0 + 3];
    const float* ib  = (const float*)d_in[i0 + 5];
    const float* jb  = (const float*)d_in[i0 + 7];
    const float* isb = (const float*)d_in[i0 + 9];
    const float* jsb = (const float*)d_in[i0 + 11];
    const float* pb  = (const float*)d_in[i0 + 13];
    const float* sb  = (const float*)d_in[i0 + 15];
    int nblk = (int)((MC + 255) / 256);
    ln_kernel<<<M, 128, 0, stream>>>(x, n1g, n1b, xn);
    launch_gemm(ACT_NONE, OUT_F32, xn, tiw[a],  ib,  tmpA, nullptr, nullptr, nullptr, nullptr, M, CDIM, CDIM, stream);
    launch_gemm(ACT_SIG,  OUT_F32, xn, tisw[a], isb, tmpB, nullptr, nullptr, nullptr, nullptr, M, CDIM, CDIM, stream);
    if (a == 0) gatewrite_kernel<false><<<nblk, 256, 0, stream>>>(tmpA, tmpB, mask, buf2);
    else        gatewrite_kernel<true ><<<nblk, 256, 0, stream>>>(tmpA, tmpB, mask, buf2);
    launch_gemm(ACT_NONE, OUT_F32, xn, tjw[a],  jb,  tmpA, nullptr, nullptr, nullptr, nullptr, M, CDIM, CDIM, stream);
    launch_gemm(ACT_SIG,  OUT_F32, xn, tjsw[a], jsb, tmpB, nullptr, nullptr, nullptr, nullptr, M, CDIM, CDIM, stream);
    if (a == 0) gatewrite_kernel<false><<<nblk, 256, 0, stream>>>(tmpA, tmpB, mask, buf3);
    else        gatewrite_kernel<true ><<<nblk, 256, 0, stream>>>(tmpA, tmpB, mask, buf3);
    launch_gemm(ACT_SIG, OUT_F32, xn, tsw[a], sb, tmpB, nullptr, nullptr, nullptr, nullptr, M, CDIM, CDIM, stream);
    trimul_kernel<<<BDIM * CDIM, 128, 0, stream>>>(buf2, buf3, tmpA);
    ln_kernel<<<M, 128, 0, stream>>>(tmpA, n2g, n2b, buf2);
    launch_gemm(ACT_NONE, OUT_RESID, buf2, tpw[a], pb, nullptr, nullptr, x, tmpB, mask, M, CDIM, CDIM, stream);
  }

  // ================= transition =================
  {
    const float* ng = (const float*)d_in[54];
    const float* nb = (const float*)d_in[55];
    const float* b1 = (const float*)d_in[57];
    const float* b2 = (const float*)d_in[59];
    ln_kernel<<<M, 128, 0, stream>>>(x, ng, nb, xn);
    launch_gemm(ACT_RELU, OUT_BF16, xn, w1t, b1, nullptr, h1, nullptr, nullptr, nullptr, M, CDIM * 4, CDIM, stream);
    launch_gemm(ACT_NONE, OUT_RESID, h1, w2t, b2, nullptr, nullptr, x, nullptr, nullptr, M, CDIM, CDIM * 4, stream);
  }

  // final x * mask
  maskmul_kernel<<<(int)((MC + 255) / 256), 256, 0, stream>>>(x, mask);
}
